// VectorQuantizer_31945966748173
// MI455X (gfx1250) — matrix operand order repaired and checked by reading, not compiled, not run on hardware
//
#include <hip/hip_runtime.h>
#include <hip/hip_bf16.h>

typedef __attribute__((ext_vector_type(16))) __bf16       v16bf;
typedef __attribute__((ext_vector_type(8)))  float        v8f;
typedef __attribute__((ext_vector_type(4)))  unsigned int u32x4;
typedef __attribute__((ext_vector_type(8)))  int          i32x8;
typedef __attribute__((ext_vector_type(4)))  int          i32x4;
typedef __attribute__((ext_vector_type(4)))  float        f32x4;

#define K_CODES 1024
#define DIM     64
#define PADH    72                 // bf16 elements per padded LDS row (144 B, stride 36 dwords)
#define PADF    65                 // f32 elements per padded LDS row for z tile
#define TOK_PER_BLK 128
#define THREADS 256
#define N_TOK   65536
#define OUT_ZQ_ELEMS 4194304       // 16*64*64*64

// LDS byte offsets (dynamic LDS only -> codebook sits at wave LDS offset 0)
#define OFF_CB   0
#define OFF_ZB   (K_CODES * PADH * 2)                  // 147456
#define OFF_ZF   (OFF_ZB + TOK_PER_BLK * PADH * 2)     // 165888
#define OFF_C2   (OFF_ZF + TOK_PER_BLK * PADF * 4)     // 199168
#define OFF_IDX  (OFF_C2 + K_CODES * 4)                // 203264
#define OFF_RED  (OFF_IDX + TOK_PER_BLK * 4)           // 203776
#define LDS_BYTES (OFF_RED + 64)                       // 203840

#if __has_builtin(__builtin_amdgcn_tensor_load_to_lds)
#define VQ_USE_TDM 1
#else
#define VQ_USE_TDM 0
#endif

union OpReg { v16bf v; u32x4 q[2]; };

// ---------- prep: codebook f32 -> bf16, per-code squared norms, zero loss ----------
__global__ __launch_bounds__(256, 1)
void vq_prep(const float* __restrict__ cb, __bf16* __restrict__ cbh,
             float* __restrict__ c2, float* __restrict__ lossSlot) {
    int k = blockIdx.x * blockDim.x + threadIdx.x;   // one thread per code
    if (k == 0) *lossSlot = 0.0f;
    if (k < K_CODES) {
        const float* row = cb + k * DIM;
        float s = 0.0f;
        #pragma unroll
        for (int d = 0; d < DIM; ++d) {
            float v = row[d];
            s = fmaf(v, v, s);
            cbh[k * DIM + d] = (__bf16)v;
        }
        c2[k] = s;
    }
}

// ---------- main: distances via bf16 WMMA, argmin, gather epilogue ----------
__global__ __launch_bounds__(THREADS, 1)
void vq_main(const float* __restrict__ z, const float* __restrict__ cbf,
             const __bf16* __restrict__ cbh, const float* __restrict__ c2g,
             float* __restrict__ outZ, float* __restrict__ lossSlot,
             float* __restrict__ outIdx) {
    extern __shared__ char smem[];
    __bf16* s_cb  = (__bf16*)(smem + OFF_CB);
    __bf16* s_zb  = (__bf16*)(smem + OFF_ZB);
    float*  s_zf  = (float*) (smem + OFF_ZF);
    float*  s_c2  = (float*) (smem + OFF_C2);
    int*    s_idx = (int*)   (smem + OFF_IDX);
    float*  s_red = (float*) (smem + OFF_RED);

    const int tid  = threadIdx.x;
    const int lane = tid & 31;
    const int wave = tid >> 5;
    const int nb   = blockIdx.x * TOK_PER_BLK;       // first token of this block
    const int bb   = nb >> 12;                       // batch index (4096 tokens per image)
    const int hw0  = nb & 4095;                      // h*64+w offset within plane
    const float* zplane = z + (size_t)bb * (DIM * 4096);

#if VQ_USE_TDM
    // One wave issues a single TDM op: 1024x64 bf16 tile, LDS rows padded by 16B
    // (pad_interval=32 dwords -> code 4, pad_amount=4 dwords -> code 3) => 144B LDS stride.
    if (wave == 0) {
        unsigned long long ga = (unsigned long long)(size_t)cbh;
        u32x4 g0; i32x8 g1;
        i32x4 gz4 = {0, 0, 0, 0};
        i32x8 gz8 = {0, 0, 0, 0, 0, 0, 0, 0};
        g0[0] = 1u;                                       // count=1, user descriptor
        g0[1] = (unsigned)OFF_CB;                         // lds_addr
        g0[2] = (unsigned)(ga & 0xffffffffull);           // global_addr[31:0]
        g0[3] = (unsigned)((ga >> 32) & 0x1ffffffull) | (2u << 30);  // addr[56:32] | type=2
        g1[0] = (1 << 16) | (1 << 20) | (4 << 22) | (3 << 25);  // 2B data, pad_en, intv=32dw, amt=4dw
        g1[1] = (DIM << 16);                              // tensor_dim0[15:0]
        g1[2] = (K_CODES << 16);                          // tensor_dim1[15:0]
        g1[3] = (DIM << 16);                              // tile_dim0
        g1[4] = K_CODES;                                  // tile_dim1
        g1[5] = DIM;                                      // tensor_dim0_stride[31:0]
        g1[6] = 0; g1[7] = 0;
        __builtin_amdgcn_tensor_load_to_lds(g0, g1, gz4, gz4, gz8, 0);
    }
#else
    // Fallback: cooperative copy with padding, 8 bf16 (16B) per op
    #pragma unroll 4
    for (int i = 0; i < (K_CODES * DIM / 8) / THREADS; ++i) {
        int e8 = i * THREADS + tid;                  // 8-element group id
        int r  = e8 >> 3, c8 = e8 & 7;
        *(u32x4*)(s_cb + r * PADH + c8 * 8) = *(const u32x4*)(cbh + e8 * 8);
    }
#endif

    // code norms -> LDS (1024 f32 = 256 float4)
    ((f32x4*)s_c2)[tid] = ((const f32x4*)c2g)[tid];

    // Stage z tile: 128 tokens x 64 dims; transpose C-major -> token-major rows.
    #pragma unroll
    for (int i = 0; i < 8; ++i) {
        int fid = i * THREADS + tid;                 // 0..2047 float4 units
        int d   = fid >> 5;                          // 0..63
        int w4  = fid & 31;
        f32x4 v = *(const f32x4*)(zplane + (size_t)d * 4096 + hw0 + w4 * 4);
        #pragma unroll
        for (int j = 0; j < 4; ++j) {
            int w = w4 * 4 + j;
            float f = v[j];
            s_zf[w * PADF + d] = f;
            s_zb[w * PADH + d] = (__bf16)f;
        }
    }

#if VQ_USE_TDM
    if (wave == 0) __builtin_amdgcn_s_wait_tensorcnt(0);
#endif
    __syncthreads();

    // ---- per-wave WMMA argmin over 1024 codes for 16 tokens ----
    const int tokOff = wave * 16;
    const int hh  = lane >> 4;                       // lane half
    const int l15 = lane & 15;

    // A operand (16x32 bf16 x2 k-steps), ISA lane layout:
    // lane<16: K0-7 in q[0], K16-23 in q[1]; lane>=16: +8 element offset.
    OpReg A0, A1;
    {
        const __bf16* ab = s_zb + (tokOff + l15) * PADH + hh * 8;
        A0.q[0] = *(const u32x4*)(ab);
        A0.q[1] = *(const u32x4*)(ab + 16);
        A1.q[0] = *(const u32x4*)(ab + 32);
        A1.q[1] = *(const u32x4*)(ab + 48);
    }

    float best[8]; int bidx[8];
    #pragma unroll
    for (int i = 0; i < 8; ++i) { best[i] = 3.4e38f; bidx[i] = 0; }

    // Unroll x2: lets the scheduler fill the WMMA->VALU hazard slots of tile t
    // with tile t+1's ds_load_b128s / WMMAs instead of v_nops.
    #pragma unroll 2
    for (int t = 0; t < K_CODES / 16; ++t) {
        int code = t * 16 + l15;
        // B operand (32x16): lane<16 holds K0-7,K16-23 of column `code`, lane>=16 K8-15,K24-31.
        const __bf16* bp = s_cb + code * PADH + hh * 8;
        OpReg B0, B1;
        B0.q[0] = *(const u32x4*)(bp);
        B0.q[1] = *(const u32x4*)(bp + 16);
        B1.q[0] = *(const u32x4*)(bp + 32);
        B1.q[1] = *(const u32x4*)(bp + 48);

        v8f acc = {};
        acc = __builtin_amdgcn_wmma_f32_16x16x32_bf16(false, A0.v, false, B0.v,
                                                      (short)0, acc, false, false);
        acc = __builtin_amdgcn_wmma_f32_16x16x32_bf16(false, A1.v, false, B1.v,
                                                      (short)0, acc, false, false);
        float c2v = s_c2[code];
        #pragma unroll
        for (int i = 0; i < 8; ++i) {
            float s = fmaf(-2.0f, acc[i], c2v);      // ||c||^2 - 2 z.c
            if (s < best[i]) { best[i] = s; bidx[i] = code; }
        }
    }

    // Reduce across the 16 column lanes (first-min-index tie break).
    #pragma unroll
    for (int i = 0; i < 8; ++i) {
        float bv = best[i]; int bi = bidx[i];
        #pragma unroll
        for (int m = 1; m < 16; m <<= 1) {
            float ob = __shfl_xor(bv, m, 16);
            int   oi = __shfl_xor(bi, m, 16);
            if (ob < bv || (ob == bv && oi < bi)) { bv = ob; bi = oi; }
        }
        if (l15 == i) s_idx[tokOff + (hh ? i + 8 : i)] = bi;  // rows 0-7 / 8-15
    }
    __syncthreads();

    // ---- epilogue: gather f32 codebook rows, NCHW scatter, fused loss ----
    float la = 0.0f;
    #pragma unroll 4
    for (int i = 0; i < 32; ++i) {
        int lin = i * THREADS + tid;                 // 0..8191
        int tok = lin & 127;
        int d   = lin >> 7;
        int ci  = s_idx[tok];
        float v = cbf[ci * DIM + d];
        float diff = v - s_zf[tok * PADF + d];
        la = fmaf(diff, diff, la);
        outZ[(size_t)bb * (DIM * 4096) + (size_t)d * 4096 + hw0 + tok] = v;
    }
    if (tid < TOK_PER_BLK) outIdx[nb + tid] = (float)s_idx[tid];

    // loss reduction: wave shuffle -> LDS -> one atomic per block
    #pragma unroll
    for (int m = 1; m < 32; m <<= 1) la += __shfl_xor(la, m, 32);
    if (lane == 0) s_red[wave] = la;
    __syncthreads();
    if (tid == 0) {
        float s = 0.0f;
        #pragma unroll
        for (int i = 0; i < 8; ++i) s += s_red[i];
        atomicAdd(lossSlot, s * (1.25f / (float)OUT_ZQ_ELEMS));  // (1+BETA)*mean
    }
}

extern "C" void kernel_launch(void* const* d_in, const int* in_sizes, int n_in,
                              void* d_out, int out_size, void* d_ws, size_t ws_size,
                              hipStream_t stream) {
    (void)in_sizes; (void)n_in; (void)out_size; (void)ws_size;
    const float* z  = (const float*)d_in[0];         // [16,64,64,64] f32
    const float* cb = (const float*)d_in[1];         // [1024,64] f32

    __bf16* cbh = (__bf16*)d_ws;                             // 128 KB
    float*  c2  = (float*)((char*)d_ws + K_CODES * DIM * 2); // 4 KB

    float* outZ     = (float*)d_out;                 // [4194304]
    float* lossSlot = outZ + OUT_ZQ_ELEMS;           // [1]
    float* outIdx   = lossSlot + 1;                  // [65536]

    (void)hipFuncSetAttribute((const void*)vq_main,
                              hipFuncAttributeMaxDynamicSharedMemorySize, LDS_BYTES);

    vq_prep<<<4, 256, 0, stream>>>(cb, cbh, c2, lossSlot);
    vq_main<<<N_TOK / TOK_PER_BLK, THREADS, LDS_BYTES, stream>>>(
        z, cb, cbh, c2, outZ, lossSlot, outIdx);
}
